// SimpleCharRNN_32323923869903
// MI455X (gfx1250) — compile-verified, matrix operand order checked by reading, not run on hardware
//
#include <hip/hip_runtime.h>

// ---------------------------------------------------------------------------
// SimpleCharRNN on MI455X (gfx1250): persistent bf16-WMMA LSTM recurrence.
//   B=256, T=512, IN=128, H=1024, OUT=9
// Folds:
//   gates_t = x_t@W_ix^T + h_{t-1}@(W_hh + W_io@W_h2o)^T + b_eff
//             (b_eff = b_ih+b_hh+W_io@b_h2o; t==0 subtracts W_io@b_h2o term)
//   out_t   = h_cell_t @ (W_h2o@W_h2h)^T + (W_h2o@b_h2h + b_h2o)
// x is pre-permuted into WMMA-A fragment order once (prep kernel), so the
// per-step x stage is a contiguous 4KB async global->LDS copy issued one
// step ahead (double buffered, retired with s_wait_asynccnt).
// ---------------------------------------------------------------------------

typedef __attribute__((ext_vector_type(16))) __bf16 v16bf;
typedef __attribute__((ext_vector_type(8)))  float  v8f;

#define B_    256
#define T_    512
#define IN_   128
#define H_    1024
#define OUT_  9
#define INC_  (IN_ + OUT_)     // 137: W_ih row stride
#define KTOT  (IN_ + H_)       // 1152 concat K for gates GEMM
#define NKT   (KTOT / 32)      // 36 K-tiles (gates)
#define NXT_  4                // K-tiles holding x
#define NNT   256              // 4096/16 N-tiles (gates)
#define HKT   (H_ / 32)        // 32 K-tiles (h2h / h2o)
#define HNT   (H_ / 16)        // 64 N-tiles (h2h)
#define TILE  512              // bf16 elements per 16x32 fragment tile
#define XSTEP (NXT_ * TILE)    // 2048 bf16 per (batch-tile, t)

// round-to-nearest-even f32 -> bf16 (raw u16)
__device__ __forceinline__ unsigned short f2bf(float f) {
  union { float f; unsigned u; } v; v.f = f;
  unsigned r = v.u + 0x7FFFu + ((v.u >> 16) & 1u);
  return (unsigned short)(r >> 16);
}
// K-offset within a 16x32 bf16 fragment for (lane, elem j) per ISA layout:
// lanes 0-15: K = {0..7, 16..23}; lanes 16-31: K = {8..15, 24..31}
__device__ __forceinline__ int koff(int l, int j) {
  return ((j >> 3) << 4) | (((l >> 4) & 1) << 3) | (j & 7);
}
__device__ __forceinline__ float sigm(float x) { return 1.0f / (1.0f + __expf(-x)); }
__device__ __forceinline__ float tanh_f(float x) { return 2.0f / (1.0f + __expf(-2.0f * x)) - 1.0f; }

#define WMMA_BF16(A, Bm, C) \
  __builtin_amdgcn_wmma_f32_16x16x32_bf16(false, (A), false, (Bm), (short)0, (C), false, false)

// ---------------- prep: x pre-permuted to A-fragment order ------------------
// xA[bt][t][kt(4)][lane(32)][j(16)] bf16; inverse mapping is k = kt*32+koff(l,j)
__global__ void rnn1250_prep_x(const float* __restrict__ x,
                               unsigned short* __restrict__ xA) {
  long idx = (long)blockIdx.x * 1024 + threadIdx.x;
  if (idx >= (long)B_ * T_ * IN_) return;
  int within = (int)(idx & (XSTEP - 1));
  int j = within & 15, l = (within >> 4) & 31, kt = within >> 9;
  long btt = idx >> 11;
  int t  = (int)(btt % T_);
  int bt = (int)(btt / T_);
  int m = l & 15;
  int k = kt * 32 + koff(l, j);
  xA[idx] = f2bf(x[(((long)(bt * 16 + m)) * T_ + t) * IN_ + k]);
}

// ---------------- prep: fused gates weight [36 kt][256 nt][512] bf16 --------
__global__ void rnn1250_prep_gatesw(const float* __restrict__ Wih,
                                    const float* __restrict__ Whh,
                                    const float* __restrict__ Wh2o,
                                    unsigned short* __restrict__ Wg) {
  long idx = (long)blockIdx.x * 1024 + threadIdx.x;
  if (idx >= (long)NKT * NNT * TILE) return;
  int j = idx & 15;
  int l = (int)((idx >> 4) & 31);
  long tile = idx >> 9;
  int ntile = (int)(tile % NNT);
  int kt = (int)(tile / NNT);
  int n = ntile * 16 + (l & 15);          // gate column 0..4095
  int k = kt * 32 + koff(l, j);           // concat K 0..1151
  float v;
  if (k < IN_) {
    v = Wih[(long)n * INC_ + k];
  } else {
    int kh = k - IN_;
    v = Whh[(long)n * H_ + kh];
#pragma unroll
    for (int p = 0; p < OUT_; ++p)
      v += Wih[(long)n * INC_ + IN_ + p] * Wh2o[(long)p * H_ + kh];
  }
  Wg[idx] = f2bf(v);
}

// ---------------- prep: h2h weight [32 kt][64 nt][512] bf16 -----------------
__global__ void rnn1250_prep_h2hw(const float* __restrict__ Wh2h,
                                  unsigned short* __restrict__ Wt) {
  long idx = (long)blockIdx.x * 1024 + threadIdx.x;
  if (idx >= (long)HKT * HNT * TILE) return;
  int j = idx & 15;
  int l = (int)((idx >> 4) & 31);
  long tile = idx >> 9;
  int ntile = (int)(tile % HNT);
  int kt = (int)(tile / HNT);
  int n = ntile * 16 + (l & 15);
  int k = kt * 32 + koff(l, j);
  Wt[idx] = f2bf(Wh2h[(long)n * H_ + k]);
}

// ------- prep: folded out weight (W_h2o@W_h2h), 9->16 cols, [32 kt][512] ----
__global__ void rnn1250_prep_outw2(const float* __restrict__ Wh2o,
                                   const float* __restrict__ Wh2h,
                                   unsigned short* __restrict__ Wt) {
  int idx = blockIdx.x * 256 + threadIdx.x;
  if (idx >= HKT * TILE) return;
  int j = idx & 15;
  int l = (idx >> 4) & 31;
  int kt = idx >> 9;
  int n = l & 15;
  int k = kt * 32 + koff(l, j);
  float v = 0.0f;
  if (n < OUT_) {
    for (int q = 0; q < H_; ++q)
      v += Wh2o[(long)n * H_ + q] * Wh2h[(long)q * H_ + k];
  }
  Wt[idx] = f2bf(v);
}

// ---------------- prep: effective biases ------------------------------------
__global__ void rnn1250_prep_bias(const float* __restrict__ bih,
                                  const float* __restrict__ bhh,
                                  const float* __restrict__ Wih,
                                  const float* __restrict__ bh2o,
                                  float* __restrict__ beff,
                                  float* __restrict__ bcorr) {
  int n = blockIdx.x * 256 + threadIdx.x;
  if (n >= 4 * H_) return;
  float c = 0.0f;
#pragma unroll
  for (int p = 0; p < OUT_; ++p) c += Wih[(long)n * INC_ + IN_ + p] * bh2o[p];
  bcorr[n] = c;
  beff[n] = bih[n] + bhh[n] + c;
}

// ---------------- prep: folded output bias (W_h2o@b_h2h + b_h2o) ------------
__global__ void rnn1250_prep_bias2(const float* __restrict__ Wh2o,
                                   const float* __restrict__ bh2h,
                                   const float* __restrict__ bh2o,
                                   float* __restrict__ bho2) {
  int n = threadIdx.x;
  if (n >= 16) return;
  float v = 0.0f;
  if (n < OUT_) {
    v = bh2o[n];
    for (int q = 0; q < H_; ++q) v += Wh2o[(long)n * H_ + q] * bh2h[q];
  }
  bho2[n] = v;
}

// ---------------- main persistent recurrence kernel -------------------------
// grid = 16 (one 16-row batch tile per WG), block = 1024 (32 waves = 1 WGP)
__global__ __launch_bounds__(1024, 1)
void rnn1250_persist(const unsigned short* __restrict__ xA,
                     const float* __restrict__ beff,
                     const float* __restrict__ bcorr,
                     const unsigned short* __restrict__ Wg,
                     const unsigned short* __restrict__ Whh2,
                     const unsigned short* __restrict__ Wout2,
                     const float* __restrict__ bh2h,
                     const float* __restrict__ bho2,
                     float* __restrict__ out) {
  __shared__ unsigned short sH[HKT * TILE];     // 32 KB: h / h_cell A-tiles
  __shared__ unsigned short sX[2 * XSTEP];      // 8 KB: double-buffered x tiles

  const int tid  = threadIdx.x;
  const int lane = tid & 31;
  const int w    = tid >> 5;        // wave 0..31
  const int ln   = lane & 15;
  const int bt   = blockIdx.x;      // batch tile: rows bt*16 .. bt*16+15

  // issue async copy of x_0 into sX[0] (waves 0..7: 256 lanes x 16B = 4KB)
  if (tid < 256) {
    const unsigned short* gsrc = xA + ((long)bt * T_) * XSTEP + tid * 8;
    unsigned ldsoff = (unsigned)(unsigned long long)&sX[tid * 8];
    asm volatile("global_load_async_to_lds_b128 %0, %1, off"
                 :: "v"(ldsoff), "v"((unsigned long long)gsrc) : "memory");
  }

  // zero initial hidden state region (h_{-1} = 0)
  for (int i = tid; i < HKT * TILE; i += 1024) sH[i] = 0;

  // per-wave tile offsets into Wg (v16bf units, without kt term): p = ht*4+g
  long noff[8];
#pragma unroll
  for (int p = 0; p < 8; ++p) {
    int ht = p >> 2, g = p & 3;
    noff[p] = (long)(g * 64 + 2 * w + ht) * 32 + lane;
  }

  // hoist per-lane biases (constant over t)
  float bg[2][4], bc[2][4], bh[2];
#pragma unroll
  for (int ht = 0; ht < 2; ++ht) {
    int htile = 2 * w + ht;
#pragma unroll
    for (int g = 0; g < 4; ++g) {
      int n = g * H_ + htile * 16 + ln;
      bg[ht][g] = beff[n];
      bc[ht][g] = bcorr[n];
    }
    bh[ht] = bh2h[htile * 16 + ln];
  }
  const float bo = (ln < OUT_) ? bho2[ln] : 0.0f;

  const v8f vzero = {0.f, 0.f, 0.f, 0.f, 0.f, 0.f, 0.f, 0.f};
  v8f cst[2];                       // register-resident cell state, all 512 steps
  cst[0] = vzero; cst[1] = vzero;

  const v16bf* sHf = (const v16bf*)sH;
  const v16bf* sXf = (const v16bf*)sX;
  const v16bf* Wgf = (const v16bf*)Wg;
  const v16bf* Whf = (const v16bf*)Whh2;
  const v16bf* Wof = (const v16bf*)Wout2;

  if (tid < 256) asm volatile("s_wait_asynccnt 0x0" ::: "memory");
  __syncthreads();

#pragma unroll 1
  for (int t = 0; t < T_; ++t) {
    const int cur = t & 1, nxt = cur ^ 1;

    // ---- issue async copy of x_{t+1} into the other buffer ----
    if (tid < 256) {
      if (t + 1 < T_) {
        const unsigned short* gsrc = xA + ((long)bt * T_ + (t + 1)) * XSTEP + tid * 8;
        unsigned ldsoff = (unsigned)(unsigned long long)&sX[nxt * XSTEP + tid * 8];
        asm volatile("global_load_async_to_lds_b128 %0, %1, off"
                     :: "v"(ldsoff), "v"((unsigned long long)gsrc) : "memory");
        asm volatile("s_wait_asynccnt 0x1" ::: "memory");  // x_t landed (in-order)
      } else {
        asm volatile("s_wait_asynccnt 0x0" ::: "memory");
      }
    }
    __syncthreads();   // x_t visible; h_prev writes from t-1 ordered

    // ---- gates GEMM: [16 x 1152] @ [1152 x 4096]-slice, 8-deep B preload ---
    v8f acc[8];
#pragma unroll
    for (int p = 0; p < 8; ++p) acc[p] = vzero;

    auto gemm8 = [&](v16bf a, long kbase) {
      v16bf b0 = Wgf[kbase + noff[0]];
      v16bf b1 = Wgf[kbase + noff[1]];
      v16bf b2 = Wgf[kbase + noff[2]];
      v16bf b3 = Wgf[kbase + noff[3]];
      v16bf b4 = Wgf[kbase + noff[4]];
      v16bf b5 = Wgf[kbase + noff[5]];
      v16bf b6 = Wgf[kbase + noff[6]];
      v16bf b7 = Wgf[kbase + noff[7]];
      // speculative prefetch of next K-tile (harmless past end: spec TH=0)
      __builtin_prefetch((const void*)&Wgf[kbase + (NNT * 32) + noff[0]], 0, 1);
      acc[0] = WMMA_BF16(a, b0, acc[0]);
      acc[1] = WMMA_BF16(a, b1, acc[1]);
      acc[2] = WMMA_BF16(a, b2, acc[2]);
      acc[3] = WMMA_BF16(a, b3, acc[3]);
      acc[4] = WMMA_BF16(a, b4, acc[4]);
      acc[5] = WMMA_BF16(a, b5, acc[5]);
      acc[6] = WMMA_BF16(a, b6, acc[6]);
      acc[7] = WMMA_BF16(a, b7, acc[7]);
    };

#pragma unroll 1
    for (int kt = 0; kt < NXT_; ++kt) {            // x part of K
      v16bf a = sXf[cur * (XSTEP / 16) + kt * 32 + lane];
      gemm8(a, (long)kt * (NNT * 32));
    }
#pragma unroll 1
    for (int kt = NXT_; kt < NKT; ++kt) {          // h part of K
      v16bf a = sHf[(kt - NXT_) * 32 + lane];
      gemm8(a, (long)kt * (NNT * 32));
    }
    __syncthreads();   // all waves done reading h_prev region

    // ---- LSTM pointwise; write h_cell (bf16) into sH ----
    float bb[2][4];
    {
      const float t0 = (t == 0) ? 1.0f : 0.0f;
#pragma unroll
      for (int ht = 0; ht < 2; ++ht)
#pragma unroll
        for (int g = 0; g < 4; ++g) bb[ht][g] = bg[ht][g] - t0 * bc[ht][g];
    }
#pragma unroll
    for (int ht = 0; ht < 2; ++ht) {
      int hidden = (2 * w + ht) * 16 + ln;
#pragma unroll
      for (int r = 0; r < 8; ++r) {
        float iv = sigm(acc[ht * 4 + 0][r] + bb[ht][0]);
        float fv = sigm(acc[ht * 4 + 1][r] + bb[ht][1]);
        float gv = tanh_f(acc[ht * 4 + 2][r] + bb[ht][2]);
        float ov = sigm(acc[ht * 4 + 3][r] + bb[ht][3]);
        float cn = fv * cst[ht][r] + iv * gv;
        cst[ht][r] = cn;
        float hc = ov * tanh_f(cn);
        int m = r + ((lane >> 4) << 3);
        int kt = hidden >> 5, kk = hidden & 31;
        int l2 = m | (((kk >> 3) & 1) << 4);
        int j2 = ((kk >> 4) << 3) | (kk & 7);
        sH[kt * TILE + l2 * 16 + j2] = f2bf(hc);
      }
    }
    __syncthreads();   // h_cell fully written

    // ---- h2h GEMM (all waves) + folded h2o (wave 0), both read h_cell ----
    v8f hacc[2];
    hacc[0] = vzero; hacc[1] = vzero;
#pragma unroll 1
    for (int kt = 0; kt < HKT; kt += 2) {
      v16bf a0  = sHf[kt * 32 + lane];
      v16bf a1  = sHf[(kt + 1) * 32 + lane];
      v16bf b00 = Whf[(long)(kt * HNT + 2 * w) * 32 + lane];
      v16bf b01 = Whf[(long)(kt * HNT + 2 * w + 1) * 32 + lane];
      v16bf b10 = Whf[(long)((kt + 1) * HNT + 2 * w) * 32 + lane];
      v16bf b11 = Whf[(long)((kt + 1) * HNT + 2 * w + 1) * 32 + lane];
      hacc[0] = WMMA_BF16(a0, b00, hacc[0]);
      hacc[1] = WMMA_BF16(a0, b01, hacc[1]);
      hacc[0] = WMMA_BF16(a1, b10, hacc[0]);
      hacc[1] = WMMA_BF16(a1, b11, hacc[1]);
    }
    if (w == 0) {
      // out_t = h_cell @ (W_h2o@W_h2h)^T + bho2  (overlaps other waves' h2h)
      v8f oacc = vzero;
#pragma unroll 1
      for (int kt = 0; kt < HKT; kt += 2) {
        v16bf a0 = sHf[kt * 32 + lane];
        v16bf a1 = sHf[(kt + 1) * 32 + lane];
        v16bf c0 = Wof[kt * 32 + lane];
        v16bf c1 = Wof[(kt + 1) * 32 + lane];
        oacc = WMMA_BF16(a0, c0, oacc);
        oacc = WMMA_BF16(a1, c1, oacc);
      }
      if (ln < OUT_) {
#pragma unroll
        for (int r = 0; r < 8; ++r) {
          int m = r + ((lane >> 4) << 3);
          out[(((long)(bt * 16 + m)) * T_ + t) * OUT_ + ln] = oacc[r] + bo;
        }
      }
    }
    __syncthreads();   // everyone done reading h_cell

    // ---- write h_new = h2h(h_cell)+b_h2h into sH for next step's A ----
#pragma unroll
    for (int i = 0; i < 2; ++i) {
      int hidden = (2 * w + i) * 16 + ln;
#pragma unroll
      for (int r = 0; r < 8; ++r) {
        float hv = hacc[i][r] + bh[i];
        int m = r + ((lane >> 4) << 3);
        int kt = hidden >> 5, kk = hidden & 31;
        int l2 = m | (((kk >> 3) & 1) << 4);
        int j2 = ((kk >> 4) << 3) | (kk & 7);
        sH[kt * TILE + l2 * 16 + j2] = f2bf(hv);
      }
    }
    // next iteration's post-stage barrier orders h_new writes vs. gates reads
  }
}

// ---------------------------------------------------------------------------
extern "C" void kernel_launch(void* const* d_in, const int* in_sizes, int n_in,
                              void* d_out, int out_size, void* d_ws, size_t ws_size,
                              hipStream_t stream) {
  const float* x    = (const float*)d_in[0];
  const float* Wih  = (const float*)d_in[1];
  const float* bih  = (const float*)d_in[2];
  const float* Whh  = (const float*)d_in[3];
  const float* bhh  = (const float*)d_in[4];
  const float* Wh2h = (const float*)d_in[5];
  const float* bh2h = (const float*)d_in[6];
  const float* Wh2o = (const float*)d_in[7];
  const float* bh2o = (const float*)d_in[8];
  const int*   lens = (const int*)d_in[9];
  float* out = (float*)d_out;

  // workspace layout (bf16 tiles, pre-permuted x, then f32 biases); ~45 MB
  unsigned short* Wg    = (unsigned short*)d_ws;                   // 36*256*512
  unsigned short* Whh2  = Wg    + (size_t)NKT * NNT * TILE;        // 32*64*512
  unsigned short* Wout2 = Whh2  + (size_t)HKT * HNT * TILE;        // 32*512
  unsigned short* xAb   = Wout2 + (size_t)HKT * TILE;              // B*T*IN bf16
  float* beff  = (float*)(xAb + (size_t)B_ * T_ * IN_);            // 4096
  float* bcorr = beff + 4 * H_;                                    // 4096
  float* bho2  = bcorr + 4 * H_;                                   // 16

  {
    long n = (long)B_ * T_ * IN_;
    rnn1250_prep_x<<<(unsigned)((n + 1023) / 1024), 1024, 0, stream>>>(x, xAb);
  }
  {
    long n = (long)NKT * NNT * TILE;
    rnn1250_prep_gatesw<<<(unsigned)((n + 1023) / 1024), 1024, 0, stream>>>(Wih, Whh, Wh2o, Wg);
  }
  {
    long n = (long)HKT * HNT * TILE;
    rnn1250_prep_h2hw<<<(unsigned)((n + 1023) / 1024), 1024, 0, stream>>>(Wh2h, Whh2);
  }
  rnn1250_prep_outw2<<<(HKT * TILE + 255) / 256, 256, 0, stream>>>(Wh2o, Wh2h, Wout2);
  rnn1250_prep_bias<<<(4 * H_ + 255) / 256, 256, 0, stream>>>(bih, bhh, Wih, bh2o, beff, bcorr);
  rnn1250_prep_bias2<<<1, 16, 0, stream>>>(Wh2o, bh2h, bh2o, bho2);

  rnn1250_persist<<<B_ / 16, 1024, 0, stream>>>(xAb, beff, bcorr, Wg, Whh2, Wout2,
                                                bh2h, bho2, out);

  // second tuple output: lengths (int32 bits appended after [B,T,9] floats)
  hipMemcpyAsync(out + (size_t)B_ * T_ * OUT_, lens, B_ * sizeof(int),
                 hipMemcpyDeviceToDevice, stream);
}
